// SimplifiedPNAAggregator_25984552141030
// MI455X (gfx1250) — matrix operand order repaired and checked by reading, not compiled, not run on hardware
//
#include <hip/hip_runtime.h>

typedef __attribute__((ext_vector_type(16))) _Float16 v16h;
typedef __attribute__((ext_vector_type(8)))  float    v8f;
typedef __attribute__((ext_vector_type(4)))  float    f32x4;

#define N_NODES 50000
#define DIMS    128
#define E_EDGES 800000
#define KTOT    640            // 5*D
#define NKT     20             // 640 / 32 WMMA K-steps
#define SH_STRIDE 648          // padded LDS row stride (halfs), even, bank-friendly

// ---- order-preserving float <-> uint key (for u32 atomicMax/Min) ----
__device__ __forceinline__ unsigned int fkey(float f) {
    unsigned int u = __float_as_uint(f);
    return (u & 0x80000000u) ? ~u : (u | 0x80000000u);
}
__device__ __forceinline__ float funkey(unsigned int k) {
    unsigned int u = (k & 0x80000000u) ? (k & 0x7FFFFFFFu) : ~k;
    return __uint_as_float(u);
}

// ---- zero/identity init of L2-resident accumulators ----
__global__ void init_ws(float* __restrict__ sum, float* __restrict__ sumsq,
                        unsigned int* __restrict__ maxk, unsigned int* __restrict__ mink,
                        float* __restrict__ cnt) {
    int i = blockIdx.x * blockDim.x + threadIdx.x;   // grid covers N*D exactly
    sum[i]   = 0.0f;
    sumsq[i] = 0.0f;
    maxk[i]  = 0u;            // < key of any real float
    mink[i]  = 0xFFFFFFFFu;   // > key of any real float
    if (i < N_NODES) cnt[i] = 0.0f;
}

// ---- pack W^T into per-lane WMMA B-fragment order, f16 ----
// bp[((j*NKT + kt)*32 + lane)*16 + e] = W[(j*16 + lane%16)*640 + kt*32 + e + 16*(lane>=16)]
__global__ void pack_w(const float* __restrict__ W, _Float16* __restrict__ bp) {
    int t = blockIdx.x * blockDim.x + threadIdx.x;   // 0 .. 8*20*32*16-1 = 81919
    int e    = t & 15;
    int lane = (t >> 4) & 31;
    int kt   = (t >> 9) % NKT;
    int j    = t / (NKT * 512);
    int K    = kt * 32 + e + ((lane >= 16) ? 16 : 0);
    int col  = j * 16 + (lane & 15);
    bp[t] = (_Float16)W[col * KTOT + K];
}

// ---- scatter: one wave per edge (32 lanes x float4 = 128 dims), NT stream ----
__global__ void scatter(const float* __restrict__ msg, const int* __restrict__ dst,
                        float* __restrict__ sum, float* __restrict__ sumsq,
                        unsigned int* __restrict__ maxk, unsigned int* __restrict__ mink,
                        float* __restrict__ cnt) {
    unsigned int g = blockIdx.x * blockDim.x + threadIdx.x;  // < E*32 exactly
    int e = (int)(g >> 5);
    int c = (int)(g & 31);
    int d = dst[e];
    const f32x4* p = (const f32x4*)(msg + (size_t)e * DIMS + c * 4);
    f32x4 m = __builtin_nontemporal_load(p);   // don't pollute L2 with the 410MB stream
    int base = d * DIMS + c * 4;
#pragma unroll
    for (int k = 0; k < 4; ++k) {
        float v = m[k];
        atomicAdd(&sum[base + k], v);
        atomicAdd(&sumsq[base + k], v * v);
        atomicMax(&maxk[base + k], fkey(v));
        atomicMin(&mink[base + k], fkey(v));
    }
    if (c == 0) atomicAdd(&cnt[d], 1.0f);
}

// ---- fused: build f16 agg tile in LDS, then WMMA GEMM + bias ----
__global__ void __launch_bounds__(256) pna_gemm(
    const float* __restrict__ sum, const float* __restrict__ sumsq,
    const unsigned int* __restrict__ maxk, const unsigned int* __restrict__ mink,
    const float* __restrict__ cnt, const float* __restrict__ h_prev,
    const _Float16* __restrict__ bp, const float* __restrict__ bias,
    float* __restrict__ out) {
    __shared__ _Float16 sh[16 * SH_STRIDE];
    const int tid   = threadIdx.x;
    const int nbase = blockIdx.x * 16;           // N = 3125*16 exactly

    // Stage 1: 16 nodes x 640 features -> f16 in LDS
    for (int idx = tid; idx < 16 * KTOT; idx += 256) {
        int nr = idx / KTOT;
        int k  = idx - nr * KTOT;
        int n  = nbase + nr;
        int reg = k >> 7;        // 0:mean 1:max 2:min 3:std 4:h_prev
        int kk  = k & 127;
        float val;
        if (reg == 4) {
            val = h_prev[n * DIMS + kk];
        } else {
            float c  = cnt[n];
            int   si = n * DIMS + kk;
            if (reg == 0 || reg == 3) {
                float cc   = fmaxf(c, 1.0f);
                float mean = sum[si] / cc;
                if (reg == 0) {
                    val = mean;
                } else {
                    float msq = sumsq[si] / cc;
                    val = sqrtf(fmaxf(msq - mean * mean, 1e-6f));
                }
            } else if (reg == 1) {
                val = (c > 0.0f) ? funkey(maxk[si]) : 0.0f;
            } else {
                val = (c > 0.0f) ? funkey(mink[si]) : 0.0f;
            }
        }
        sh[nr * SH_STRIDE + k] = (_Float16)val;
    }
    __syncthreads();

    // Stage 2: 8 waves, wave j -> 16x16 output tile (rows nbase..+15, cols j*16..+15)
    const int wave = tid >> 5;      // column tile j
    const int lane = tid & 31;
    const int m16  = lane & 15;     // A row / D column index
    const int hi   = lane >> 4;     // lane group

    float bj = bias[wave * 16 + m16];
    v8f c;
#pragma unroll
    for (int r = 0; r < 8; ++r) c[r] = bj;

    const unsigned int* shw = (const unsigned int*)sh;
    const int arow = m16 * (SH_STRIDE / 2);     // u32 units

    for (int kt = 0; kt < NKT; ++kt) {
        union { v16h v; unsigned int u[8]; } a;
        const int k0 = kt * 32;
#pragma unroll
        for (int i = 0; i < 8; ++i) {
            // A fragment K pairs: {0,1},{2,3},...,{14,15} (+16 if hi)
            int Kp = 2 * i + hi * 16;
            a.u[i] = shw[arow + ((k0 + Kp) >> 1)];
        }
        v16h bvec = *(const v16h*)(bp + ((((wave * NKT + kt) * 32) + lane) << 4));
        c = __builtin_amdgcn_wmma_f32_16x16x32_f16(
                /*neg_a=*/false, a.v, /*neg_b=*/false, bvec,
                /*c_mod=*/(short)0, c, /*reuse_a=*/false, /*reuse_b=*/false);
    }

#pragma unroll
    for (int r = 0; r < 8; ++r) {
        int M = r + hi * 8;
        out[(size_t)(nbase + M) * DIMS + wave * 16 + m16] = c[r];
    }
}

extern "C" void kernel_launch(void* const* d_in, const int* in_sizes, int n_in,
                              void* d_out, int out_size, void* d_ws, size_t ws_size,
                              hipStream_t stream) {
    (void)in_sizes; (void)n_in; (void)out_size; (void)ws_size;
    const float* msg    = (const float*)d_in[0];
    const int*   dst    = (const int*)d_in[1];
    const float* h_prev = (const float*)d_in[2];
    const float* W      = (const float*)d_in[3];
    const float* bias   = (const float*)d_in[4];
    float* out = (float*)d_out;

    char* ws = (char*)d_ws;
    const size_t nd = (size_t)N_NODES * DIMS;            // 6.4M elems, 25.6MB each
    float*        sum   = (float*)ws;        ws += nd * 4;
    float*        sumsq = (float*)ws;        ws += nd * 4;
    unsigned int* maxk  = (unsigned int*)ws; ws += nd * 4;
    unsigned int* mink  = (unsigned int*)ws; ws += nd * 4;
    float*        cnt   = (float*)ws;        ws += (((size_t)N_NODES * 4) + 255) / 256 * 256;
    _Float16*     bp    = (_Float16*)ws;     // 8*20*32*16 f16 = 160KB

    init_ws<<<(unsigned)(nd / 256), 256, 0, stream>>>(sum, sumsq, maxk, mink, cnt);
    pack_w<<<(8 * NKT * 32 * 16) / 256, 256, 0, stream>>>(W, bp);
    scatter<<<(unsigned)((size_t)E_EDGES * 32 / 256), 256, 0, stream>>>(
        msg, dst, sum, sumsq, maxk, mink, cnt);
    pna_gemm<<<N_NODES / 16, 256, 0, stream>>>(
        sum, sumsq, maxk, mink, cnt, h_prev, bp, bias, out);
}
